// min_model_85830626443605
// MI455X (gfx1250) — matrix operand order repaired and checked by reading, not compiled, not run on hardware
//
#include <hip/hip_runtime.h>

// Problem constants (from reference): B=16, T=2048, V=512
#define B_ 16
#define T_ 2048
#define V_ 512

typedef __attribute__((ext_vector_type(16))) _Float16 v16h;
typedef __attribute__((ext_vector_type(8)))  float    v8f;

union HFrag {
    v16h h;
    unsigned int u[8];
    _Float16 f[16];
};

// CDNA5 async global->LDS copy: INST_OFFSET is added to BOTH the LDS and the
// global address, so one base pair + immediate offsets covers a whole chunk.
// Tracked by ASYNCcnt (s_wait_asynccnt).
#define ASYNC_CP16(ldsoff, gaddr, off)                                        \
    asm volatile("global_load_async_to_lds_b128 %0, %1, off offset:" #off     \
                 :: "v"(ldsoff), "v"(gaddr) : "memory")
#define ASYNC_WAIT()                                                          \
    asm volatile("s_wait_asynccnt 0x0" ::: "memory")

// Low 32 bits of a generic pointer to LDS == LDS byte offset (aperture keeps
// the LDS offset in addr[31:0]).
__device__ __forceinline__ unsigned lds_off(const void* p) {
    return (unsigned)(size_t)p;
}

// ---------------------------------------------------------------------------
// Kernel 1: transpose + f32->f16 convert Wq and Wv.
// WqT[c][j] = (f16) Wq[j][c]  so that Q rows (= Wq columns) are contiguous.
// ---------------------------------------------------------------------------
__global__ void prep_transpose(const float* __restrict__ Wq,
                               const float* __restrict__ Wv,
                               _Float16* __restrict__ WqT,
                               _Float16* __restrict__ WvT) {
    int id = blockIdx.x * blockDim.x + threadIdx.x;   // 0 .. 2*V*V-1
    const float* src = (id < V_ * V_) ? Wq : Wv;
    _Float16*    dst = (id < V_ * V_) ? WqT : WvT;
    int e = id & (V_ * V_ - 1);
    int c = e >> 9;        // output row = source column
    int j = e & (V_ - 1);  // output col = source row
    dst[c * V_ + j] = (_Float16)src[j * V_ + c];
}

// ---------------------------------------------------------------------------
// Kernel 2: l1[b,t,v] = sum_{s<=t, idx[b,s]==v} pos_table[t-s]  (scatter-add)
// One block per (b,t); LDS f32 accumulator with ds_add_f32 atomics.
// ---------------------------------------------------------------------------
__global__ void build_l1(const int* __restrict__ idx,
                         const float* __restrict__ pos_table,
                         _Float16* __restrict__ l1h) {
    __shared__ float acc[V_];
    const int bt  = blockIdx.x;          // b*T + t
    const int b   = bt >> 11;
    const int t   = bt & (T_ - 1);
    const int tid = threadIdx.x;

    acc[tid]       = 0.0f;
    acc[tid + 256] = 0.0f;
    __syncthreads();

    const int* idr = idx + b * T_;
    for (int s = tid; s <= t; s += 256) {
        atomicAdd(&acc[idr[s]], pos_table[t - s]);   // -> ds_add_f32
    }
    __syncthreads();

    _Float16* o = l1h + (size_t)bt * V_;
    o[tid]       = (_Float16)acc[tid];
    o[tid + 256] = (_Float16)acc[tid + 256];
}

// ---------------------------------------------------------------------------
// Kernel 3: fused causal attention (no softmax):
//   out[b,t,:] = sum_{s<=t} (Q[b,t]·l1[b,s]) * WvT[idx[b,s],:]
// Block = one (b, 16-row query tile), 8 wave32 waves, 256 threads.
// Per 32-key block: waves 0..1 compute S = Q @ K^T (two 16x16 tiles, K=512),
// mask causally, park S in LDS as f16; all 8 waves then do PV with S as the
// A matrix (K = 32 keys) into 4 persistent f32 accumulators per wave.
// Key/value staging uses CDNA5 async global->LDS copies (ASYNCcnt).
// ---------------------------------------------------------------------------
#define SKS 520   // sK row stride in f16 (512 + pad, keeps 16B alignment)
#define SVS 264   // sV row stride in f16 (256 + pad)
#define SSS 34    // sS row stride in f16 (32 + pad)

__global__ void __launch_bounds__(256, 1)
attn_kernel(const int* __restrict__ idx,
            const _Float16* __restrict__ WqT,
            const _Float16* __restrict__ WvT,
            const _Float16* __restrict__ l1h,
            float* __restrict__ out) {
    __shared__ __align__(16) _Float16 sK[32 * SKS];   // key block: 32 x 512
    __shared__ __align__(16) _Float16 sV[32 * SVS];   // value half: 32 x 256
    __shared__ __align__(16) _Float16 sS[16 * SSS];   // scores:    16 x 32

    const int tid  = threadIdx.x;
    const int lane = tid & 31;
    const int w    = tid >> 5;          // wave id 0..7
    const int ln16 = lane & 15;
    const int lhi  = (lane >> 4) & 1;   // lane-half select
    const int q0   = blockIdx.x * 16;   // query tile base row
    const int b    = blockIdx.y;
    const int bT   = b * T_;

    const v8f vzero = {0.f, 0.f, 0.f, 0.f, 0.f, 0.f, 0.f, 0.f};
    v8f acc[4];                          // wave's 4 output tiles (16x16 f32)
#pragma unroll
    for (int i = 0; i < 4; ++i) acc[i] = vzero;

    const int nkb = ((q0 + 15) >> 5) + 1;   // key blocks needed (causal)

    // Per-thread staging coordinates (reused every key block)
    const int sr  = tid >> 3;            // staged key row 0..31
    const int kjb = (tid & 7) * 64;      // sK chunk base (f16)
    const int vjb = (tid & 7) * 32;      // sV chunk base (f16)
    const unsigned ldsK = lds_off(sK + sr * SKS + kjb);
    const unsigned ldsV = lds_off(sV + sr * SVS + vjb);

    // Per-lane Q row (A-matrix row m = ln16): column idx of Wq == row of WqT.
    const unsigned int* qrow32 = (const unsigned int*)WqT;
    if (w < 2) {
        int col = idx[bT + q0 + ln16];
        qrow32 = (const unsigned int*)(WqT + (size_t)col * V_);
    }

    for (int kb = 0; kb < nkb; ++kb) {
        const int s0 = kb * 32;
        __syncthreads();
        // ---- async-stage key block: sK[r][0..511] = l1[b, s0+r, :] ----
        {
            const _Float16* g = l1h + ((size_t)(bT + s0 + sr)) * V_ + kjb;
            ASYNC_CP16(ldsK, g, 0);
            ASYNC_CP16(ldsK, g, 16);
            ASYNC_CP16(ldsK, g, 32);
            ASYNC_CP16(ldsK, g, 48);
            ASYNC_CP16(ldsK, g, 64);
            ASYNC_CP16(ldsK, g, 80);
            ASYNC_CP16(ldsK, g, 96);
            ASYNC_CP16(ldsK, g, 112);
            // prefetch next key block's l1 rows while this one is consumed
            if (s0 + 32 + sr < T_)
                __builtin_prefetch(l1h + ((size_t)(bT + s0 + 32 + sr)) * V_ + kjb,
                                   0, 1);
            ASYNC_WAIT();
        }
        __syncthreads();

        // ---- GEMM1: waves 0,1 each compute one 16x16 S subtile ----
        if (w < 2) {
            v8f sacc = vzero;
            const unsigned int* krow32 =
                (const unsigned int*)(sK + (w * 16 + ln16) * SKS);
            const int klo2 = lhi * 8;  // contiguous k order: lanes 16-31 take K+16
#pragma unroll
            for (int f = 0; f < 16; ++f) {           // K = 512 = 16 * 32
                HFrag a, bf;
#pragma unroll
                for (int v = 0; v < 8; ++v) {
                    // A operand, contiguous k order used here: k = 16*lhi + 2*v (+1)
                    int k = f * 32 + lhi * 16 + v * 2;
                    a.u[v]  = qrow32[k >> 1];
                    // B operand, same k order as A: k = 16*lhi + 2*v (+1), n = ln16
                    bf.u[v] = krow32[f * 16 + klo2 + v];
                }
                sacc = __builtin_amdgcn_wmma_f32_16x16x32_f16(
                    false, a.h, false, bf.h, (short)0, sacc, false, false);
            }
            // causal mask in C layout, convert to f16, park in LDS
#pragma unroll
            for (int r = 0; r < 8; ++r) {
                int m  = r + 8 * lhi;                // C: row = r (+8 hi half)
                int sg = s0 + w * 16 + ln16;         // global key index
                float vv = (sg <= q0 + m) ? sacc[r] : 0.0f;
                sS[m * SSS + w * 16 + ln16] = (_Float16)vv;
            }
        }

        // ---- GEMM2 (PV) in two 256-column halves of the V dimension ----
        for (int h = 0; h < 2; ++h) {
            __syncthreads();
            // async-stage value half: sV[r][0..255] = WvT[idx[b,s0+r]][h*256..]
            {
                int col = idx[bT + s0 + sr];
                const _Float16* g = WvT + (size_t)col * V_ + h * 256 + vjb;
                ASYNC_CP16(ldsV, g, 0);
                ASYNC_CP16(ldsV, g, 16);
                ASYNC_CP16(ldsV, g, 32);
                ASYNC_CP16(ldsV, g, 48);
                ASYNC_WAIT();
            }
            __syncthreads();

            // A-fragment of S (16x32, K = 32 keys)
            HFrag sa;
            {
                const unsigned int* srow32 =
                    (const unsigned int*)(sS + ln16 * SSS);
#pragma unroll
                for (int v = 0; v < 8; ++v) {
                    int k = lhi * 16 + v * 2;
                    sa.u[v] = srow32[k >> 1];
                }
            }
            const int klo = lhi * 16;
#pragma unroll
            for (int jtl = 0; jtl < 2; ++jtl) {      // wave's 2 tiles / half
                int jcol = (w * 2 + jtl) * 16 + ln16;  // col within half
                HFrag bf;
#pragma unroll
                for (int v = 0; v < 8; ++v) {
                    int k0 = klo + 2 * v;
                    bf.f[2 * v]     = sV[k0 * SVS + jcol];
                    bf.f[2 * v + 1] = sV[(k0 + 1) * SVS + jcol];
                }
                acc[h * 2 + jtl] = __builtin_amdgcn_wmma_f32_16x16x32_f16(
                    false, sa.h, false, bf.h, (short)0, acc[h * 2 + jtl],
                    false, false);
            }
        }
    }

    // ---- store output (f32), C-layout: VGPR r -> row r (+8 for hi lanes) ----
#pragma unroll
    for (int h = 0; h < 2; ++h) {
#pragma unroll
        for (int jtl = 0; jtl < 2; ++jtl) {
            int col = h * 256 + (w * 2 + jtl) * 16 + ln16;
#pragma unroll
            for (int r = 0; r < 8; ++r) {
                int row = q0 + r + 8 * lhi;
                out[((size_t)(bT + row)) * V_ + col] = acc[h * 2 + jtl][r];
            }
        }
    }
}

// ---------------------------------------------------------------------------
// Host launch
// ---------------------------------------------------------------------------
extern "C" void kernel_launch(void* const* d_in, const int* in_sizes, int n_in,
                              void* d_out, int out_size, void* d_ws, size_t ws_size,
                              hipStream_t stream) {
    (void)in_sizes; (void)n_in; (void)out_size; (void)ws_size;

    const int*   idx = (const int*)d_in[0];     // (B,T) int32
    const float* pos = (const float*)d_in[1];   // (T,1) f32
    const float* Wq  = (const float*)d_in[2];   // (V,V) f32
    const float* Wv  = (const float*)d_in[3];   // (V,V) f32
    float* out = (float*)d_out;                 // (B,T,V) f32

    // Workspace layout: WqT (512KB) | WvT (512KB) | l1h (32MB)
    char* ws = (char*)d_ws;
    _Float16* WqT = (_Float16*)(ws);
    _Float16* WvT = (_Float16*)(ws + 512 * 1024);
    _Float16* l1h = (_Float16*)(ws + 1024 * 1024);

    prep_transpose<<<2 * V_ * V_ / 256, 256, 0, stream>>>(Wq, Wv, WqT, WvT);
    build_l1<<<B_ * T_, 256, 0, stream>>>(idx, pos, l1h);

    dim3 grid(T_ / 16, B_);
    attn_kernel<<<grid, 256, 0, stream>>>(idx, WqT, WvT, l1h, out);
}
